// AttentionBlock_44813688766689
// MI455X (gfx1250) — matrix operand order repaired and checked by reading, not compiled, not run on hardware
//
#include <hip/hip_runtime.h>

// ---------------------------------------------------------------------------
// GPT-2 attention block for MI455X (gfx1250, wave32, WMMA).
// All matmuls use v_wmma_f32_16x16x32_bf16 (f32 accumulate).
// Softmax runs in base-2: 1/sqrt(D)*log2(e) is folded into Q at the QKV GEMM
// epilogue, so the attention kernel uses bare v_exp_f32 (exp2).
// ---------------------------------------------------------------------------

typedef __attribute__((ext_vector_type(16))) __bf16 bf16x16;
typedef __attribute__((ext_vector_type(8)))  __bf16 bf16x8;
typedef __attribute__((ext_vector_type(8)))  float  f32x8;

#define N_HEAD   12
#define HEAD_DIM 64
#define EMB      768
#define SEQ      2048
#define BATCH    4

// 1/sqrt(64) * log2(e), folded into Q so attention softmax is base-2
#define QSCALE   0.18033688011112042f

// f32 -> bf16 with round-to-nearest-even
__device__ __forceinline__ unsigned short f2bf(float f) {
  union { float f; unsigned u; } v; v.f = f;
  unsigned u = v.u;
  unsigned r = u + 0x7FFFu + ((u >> 16) & 1u);
  return (unsigned short)(r >> 16);
}

__device__ __forceinline__ bf16x16 cat8(bf16x8 lo, bf16x8 hi) {
  bf16x16 r;
#pragma unroll
  for (int i = 0; i < 8; ++i) { r[i] = lo[i]; r[i + 8] = hi[i]; }
  return r;
}

// ---------------------------------------------------------------------------
// GEMM: C[M,N] = cvt_bf16(A[M,768]) @ cvt_bf16(W[768,N]) + bias
// Block tile 128x128, 8 waves (4 along M x 2 along N), wave tile 32x64.
// MODE 0: A = x (f32 row-major), epilogue scatters bf16 Q^T / K / V^T.
// MODE 1: A = attention output (bf16, [B, E, S] gather), epilogue -> f32 out.
// ---------------------------------------------------------------------------
template <int MODE>
__global__ __launch_bounds__(256) void gemm_kernel(
    const float* __restrict__ A32, const unsigned short* __restrict__ A16,
    const float* __restrict__ W, const float* __restrict__ bias, int N,
    unsigned short* __restrict__ Qt, unsigned short* __restrict__ Kw,
    unsigned short* __restrict__ Vt, float* __restrict__ out) {
  // Padded LDS tiles: A rows stride 40 bf16 (80B), B rows stride 136 bf16.
  __shared__ __align__(16) unsigned short As[128 * 40];
  __shared__ __align__(16) unsigned short Bs[32 * 136];

  const int tid  = threadIdx.x;
  const int lane = tid & 31;
  const int wave = tid >> 5;
  const int wm   = wave & 3;   // wave coord along M
  const int wn   = wave >> 2;  // wave coord along N
  const int n0   = blockIdx.x * 128;
  const int m0   = blockIdx.y * 128;
  const int l15  = lane & 15;
  const int hi   = lane >> 4;

  f32x8 acc[2][4];
#pragma unroll
  for (int i = 0; i < 2; ++i)
#pragma unroll
    for (int j = 0; j < 4; ++j) acc[i][j] = (f32x8)0.0f;

  for (int kt = 0; kt < EMB; kt += 32) {
    // ---- stage A tile (128 x 32) into LDS as bf16 ----
#pragma unroll
    for (int i = 0; i < 16; ++i) {
      int idx = tid + i * 256;
      if (MODE == 0) {
        int kl = idx & 31, ml = idx >> 5;  // consecutive tid -> consecutive k
        As[ml * 40 + kl] = f2bf(A32[(size_t)(m0 + ml) * EMB + kt + kl]);
      } else {
        int ml = idx & 127, kl = idx >> 7; // consecutive tid -> consecutive s
        int m = m0 + ml;
        int b = m >> 11, s = m & 2047;
        As[ml * 40 + kl] = A16[((size_t)b * EMB + kt + kl) * SEQ + s];
      }
    }
    // ---- stage W tile (32 x 128) into LDS as bf16 ----
#pragma unroll
    for (int i = 0; i < 16; ++i) {
      int idx = tid + i * 256;
      int nl = idx & 127, kl = idx >> 7;
      Bs[kl * 136 + nl] = f2bf(W[(size_t)(kt + kl) * N + n0 + nl]);
    }
    __syncthreads();

    // A fragments: lane<16 holds row, k {0-7,16-23}; lane>=16 k {8-15,24-31}
    bf16x16 af[2];
#pragma unroll
    for (int mi = 0; mi < 2; ++mi) {
      int base = (wm * 32 + mi * 16 + l15) * 40 + hi * 8;
      af[mi] = cat8(*(const bf16x8*)&As[base], *(const bf16x8*)&As[base + 16]);
    }
    // B fragments: lane%16 == n column, lane<16 k {0-7,16-23}; lane>=16 k {8-15,24-31}
#pragma unroll
    for (int ni = 0; ni < 4; ++ni) {
      int base = wn * 64 + ni * 16 + l15;
      bf16x16 bfv;
#pragma unroll
      for (int s = 0; s < 16; ++s)
        bfv[s] = __builtin_bit_cast(__bf16, Bs[((s & 7) + hi * 8 + (s >> 3) * 16) * 136 + base]);
#pragma unroll
      for (int mi = 0; mi < 2; ++mi)
        acc[mi][ni] = __builtin_amdgcn_wmma_f32_16x16x32_bf16(
            false, af[mi], false, bfv, (short)0, acc[mi][ni], false, false);
    }
    __syncthreads();
  }

  // ---- epilogue ----
#pragma unroll
  for (int mi = 0; mi < 2; ++mi) {
#pragma unroll
    for (int ni = 0; ni < 4; ++ni) {
      int n  = n0 + wn * 64 + ni * 16 + l15;
      float bv = bias[n];
#pragma unroll
      for (int j = 0; j < 8; ++j) {
        int m = m0 + wm * 32 + mi * 16 + j + hi * 8;
        float v = acc[mi][ni][j] + bv;
        if (MODE == 1) {
          out[(size_t)m * EMB + n] = v;
        } else {
          int b = m >> 11, s = m & 2047;
          int sec = n / EMB;          // 0=Q, 1=K, 2=V
          int e = n - sec * EMB;
          int h = e >> 6, d = e & 63;
          size_t bh = (size_t)b * N_HEAD + h;
          if (sec == 0)      Qt[(bh * HEAD_DIM + d) * SEQ + s] = f2bf(v * QSCALE);
          else if (sec == 1) Kw[(bh * SEQ + s) * HEAD_DIM + d] = f2bf(v); // K
          else               Vt[(bh * HEAD_DIM + d) * SEQ + s] = f2bf(v); // V^T
        }
      }
    }
  }
}

// ---------------------------------------------------------------------------
// Flash attention: one wave per 16-query tile. Computes S^T = K @ Q^T so the
// softmax query-rows map to lanes (cheap reductions / per-lane rescale), then
// O^T += V^T @ P^T. Exactly one kv chunk per tile straddles the causal
// diagonal, so the steady-state loop carries no mask compares at all.
// Output bf16 [B,H,D,S].
// ---------------------------------------------------------------------------
__global__ __launch_bounds__(128) void attn_kernel(
    const unsigned short* __restrict__ Qt, const unsigned short* __restrict__ Kw,
    const unsigned short* __restrict__ Vt, unsigned short* __restrict__ Ob) {
  __shared__ __align__(16) unsigned short pbuf[4][32][16];  // per-wave P^T bounce

  const int lane = threadIdx.x & 31;
  const int w    = threadIdx.x >> 5;
  const int l15  = lane & 15;
  const int hi   = lane >> 4;
  const int q0   = (blockIdx.x * 4 + w) * 16;
  const size_t bh = (size_t)blockIdx.z * N_HEAD + blockIdx.y;

  const unsigned short* Qb = Qt + bh * HEAD_DIM * SEQ;
  const unsigned short* Kb = Kw + bh * SEQ * HEAD_DIM;
  const unsigned short* Vb = Vt + bh * HEAD_DIM * SEQ;

  // Q^T B-fragments (contraction = d): lane%16 == q column, d in table order
  bf16x16 bq[2];
#pragma unroll
  for (int g = 0; g < 2; ++g) {
#pragma unroll
    for (int s = 0; s < 16; ++s)
      bq[g][s] = __builtin_bit_cast(__bf16, Qb[(size_t)(g * 32 + (s & 7) + hi * 8 + (s >> 3) * 16) * SEQ + q0 + l15]);
  }

  f32x8 acc[4];  // O^T accumulator: 4 d-chunks x (16 d x 16 q)
#pragma unroll
  for (int i = 0; i < 4; ++i) acc[i] = (f32x8)0.0f;
  float mrun = -1e30f, lrun = 0.0f;

  auto process_chunk = [&](int kv0, bool masked) {
    // Prefetch next chunk's K / V rows into L2/L0 while we compute this one.
    __builtin_prefetch(Kb + (size_t)(kv0 + 32 + l15) * HEAD_DIM, 0, 3);
    __builtin_prefetch(Vb + (size_t)((hi * 32) + l15) * SEQ + kv0 + 32, 0, 3);

    // S^T = K @ Q^T for this 32-kv chunk (two 16x16 C frags)
    f32x8 st[2];
#pragma unroll
    for (int f = 0; f < 2; ++f) {
      st[f] = (f32x8)0.0f;
#pragma unroll
      for (int g = 0; g < 2; ++g) {
        const unsigned short* p =
            Kb + (size_t)(kv0 + f * 16 + l15) * HEAD_DIM + g * 32 + hi * 8;
        bf16x16 ak = cat8(*(const bf16x8*)p, *(const bf16x8*)(p + 16));
        st[f] = __builtin_amdgcn_wmma_f32_16x16x32_bf16(
            false, ak, false, bq[g], (short)0, st[f], false, false);
      }
    }

    // local max over kv (column q == lane); causal mask only in last chunk
    const int qg = q0 + l15;
    float mloc = -1e30f;
#pragma unroll
    for (int f = 0; f < 2; ++f)
#pragma unroll
      for (int j = 0; j < 8; ++j) {
        float s = st[f][j];
        if (masked) {
          int kvg = kv0 + f * 16 + j + hi * 8;
          s = (kvg > qg) ? -1e30f : s;
          st[f][j] = s;
        }
        mloc = fmaxf(mloc, s);
      }
    mloc = fmaxf(mloc, __shfl_xor(mloc, 16, 32));
    float mnew  = fmaxf(mrun, mloc);
    float alpha = __builtin_amdgcn_exp2f(mrun - mnew);

    float psum = 0.0f;
#pragma unroll
    for (int f = 0; f < 2; ++f)
#pragma unroll
      for (int j = 0; j < 8; ++j) {
        float p = __builtin_amdgcn_exp2f(st[f][j] - mnew);
        psum += p;
        pbuf[w][f * 16 + j + hi * 8][l15] = f2bf(p);  // P^T -> LDS (kv, q)
      }
    psum += __shfl_xor(psum, 16, 32);
    lrun = lrun * alpha + psum;
    mrun = mnew;
#pragma unroll
    for (int dc = 0; dc < 4; ++dc)
#pragma unroll
      for (int j = 0; j < 8; ++j) acc[dc][j] *= alpha;

    // P^T B-fragment: lane%16 == q column, kv in table order (same-wave DS is in-order)
    bf16x16 pb;
#pragma unroll
    for (int s = 0; s < 16; ++s)
      pb[s] = __builtin_bit_cast(__bf16, pbuf[w][(s & 7) + hi * 8 + (s >> 3) * 16][l15]);

    // O^T += V^T @ P^T
#pragma unroll
    for (int dc = 0; dc < 4; ++dc) {
      const unsigned short* p =
          Vb + (size_t)(dc * 16 + l15) * SEQ + kv0 + hi * 8;
      bf16x16 av = cat8(*(const bf16x8*)p, *(const bf16x8*)(p + 16));
      acc[dc] = __builtin_amdgcn_wmma_f32_16x16x32_bf16(
          false, av, false, pb, (short)0, acc[dc], false, false);
    }
  };

  // Chunks [0, lastChunk) are strictly below the diagonal: no masking needed.
  const int lastChunk = (q0 + 15) >> 5;
  for (int c = 0; c < lastChunk; ++c) process_chunk(c * 32, false);
  process_chunk(lastChunk * 32, true);  // only this chunk straddles diagonal

  const float inv = 1.0f / lrun;
#pragma unroll
  for (int dc = 0; dc < 4; ++dc)
#pragma unroll
    for (int j = 0; j < 8; ++j) {
      int d = dc * 16 + j + hi * 8;
      Ob[(bh * HEAD_DIM + d) * SEQ + q0 + l15] = f2bf(acc[dc][j] * inv);
    }
}

// ---------------------------------------------------------------------------
extern "C" void kernel_launch(void* const* d_in, const int* in_sizes, int n_in,
                              void* d_out, int out_size, void* d_ws, size_t ws_size,
                              hipStream_t stream) {
  (void)in_sizes; (void)n_in; (void)out_size; (void)ws_size;
  const float* x      = (const float*)d_in[0];
  // d_in[1] is the dense causal mask; handled analytically in attn_kernel.
  const float* W_attn = (const float*)d_in[2];
  const float* b_attn = (const float*)d_in[3];
  const float* W_proj = (const float*)d_in[4];
  const float* b_proj = (const float*)d_in[5];
  float* out = (float*)d_out;

  const size_t HSD = (size_t)BATCH * N_HEAD * SEQ * HEAD_DIM;  // 6,291,456
  unsigned short* Qt = (unsigned short*)d_ws;  // [B,H,D,S] bf16 (pre-scaled)
  unsigned short* Kw = Qt + HSD;               // [B,H,S,D] bf16
  unsigned short* Vt = Kw + HSD;               // [B,H,D,S] bf16
  unsigned short* Ob = Vt + HSD;               // [B,H,D,S] bf16 attn output

  gemm_kernel<0><<<dim3(2304 / 128, (BATCH * SEQ) / 128), 256, 0, stream>>>(
      x, nullptr, W_attn, b_attn, 2304, Qt, Kw, Vt, nullptr);

  attn_kernel<<<dim3(SEQ / 64, N_HEAD, BATCH), 128, 0, stream>>>(Qt, Kw, Vt, Ob);

  gemm_kernel<1><<<dim3(EMB / 128, (BATCH * SEQ) / 128), 256, 0, stream>>>(
      nullptr, Ob, W_proj, b_proj, EMB, nullptr, nullptr, nullptr, out);
}
